// SelfAttention_23270132810254
// MI455X (gfx1250) — matrix operand order repaired and checked by reading, not compiled, not run on hardware
//
#include <hip/hip_runtime.h>
#include <hip/hip_bf16.h>

#define EMB 1024
#define NH 16
#define HD 64
#define SQ 2048
#define BB 2

typedef _Float16 v16h __attribute__((ext_vector_type(16)));
typedef _Float16 v8h  __attribute__((ext_vector_type(8)));
typedef float    v8f  __attribute__((ext_vector_type(8)));
typedef float    v4f  __attribute__((ext_vector_type(4)));
typedef unsigned int v4u __attribute__((ext_vector_type(4)));
typedef int      v8i  __attribute__((ext_vector_type(8)));
typedef int      v4i  __attribute__((ext_vector_type(4)));

#define WMMA_F16(a, b, c) \
    __builtin_amdgcn_wmma_f32_16x16x32_f16(false, (a), false, (b), (short)0, (c), false, false)

// ---------------- fragment loaders (16B-aligned LDS/global reads) ----------------

// B-fragment (32x16): lane col = lane&15, element i -> K = i + 16*half.
__device__ __forceinline__ v16h load_b16(const _Float16* __restrict__ p) {
    v8h lo = *(const v8h*)(p);
    v8h hi = *(const v8h*)(p + 8);
    v16h r;
#pragma unroll
    for (int i = 0; i < 8; ++i) { r[i] = lo[i]; r[i + 8] = hi[i]; }
    return r;
}

// A-fragment (16x32): lane row = lane&15; element i -> K = i + 16*half.
__device__ __forceinline__ v16h load_a16(const _Float16* __restrict__ row, int half) {
    v8h lo = *(const v8h*)(row + 16 * half);
    v8h hi = *(const v8h*)(row + 16 * half + 8);
    v16h r;
#pragma unroll
    for (int i = 0; i < 8; ++i) { r[i] = lo[i]; r[i + 8] = hi[i]; }
    return r;
}

// ---------------- TDM 2D tile load: global f16 -> LDS, with LDS row padding ----------------
// Descriptor per CDNA5 ISA. data_size=1 (2B elements).
__device__ __forceinline__ void tdm_load_2d(
    unsigned lds_addr, const void* gptr,
    unsigned tensor_d0, unsigned tensor_d1,
    unsigned tile_d0, unsigned tile_d1,
    unsigned long long stride0,
    unsigned pad_interval_code, unsigned pad_amount_code)
{
    unsigned long long ga = (unsigned long long)(uintptr_t)gptr;
    v4u g0 = { 1u,                                  // count=1, user mode
               lds_addr,                            // LDS byte offset
               (unsigned)ga,                        // global_addr[31:0]
               (unsigned)(((ga >> 32) & 0x1FFFFFFull) | (2u << 30)) }; // addr[56:32] | type=2
    unsigned w0 = (1u << 16)                        // data_size = 1 (2 bytes)
                | (1u << 20)                        // pad_enable
                | (pad_interval_code << 22)
                | (pad_amount_code << 25);
    v8i g1;
    g1[0] = (int)w0;
    g1[1] = (int)((tensor_d0 & 0xFFFFu) << 16);                      // tensor_dim0 lo
    g1[2] = (int)((tensor_d0 >> 16) | ((tensor_d1 & 0xFFFFu) << 16));
    g1[3] = (int)((tensor_d1 >> 16) | (tile_d0 << 16));
    g1[4] = (int)(tile_d1 & 0xFFFFu);                                // tile_dim1 (tile_dim2=0)
    g1[5] = (int)(unsigned)(stride0 & 0xFFFFFFFFull);                // tensor_dim0_stride lo
    g1[6] = (int)(unsigned)(stride0 >> 32);                          // stride hi (dim1_stride=0)
    g1[7] = 0;
    v4i z4 = {0, 0, 0, 0};
#if defined(__clang_major__) && (__clang_major__ >= 23)
    v8i z8 = {0, 0, 0, 0, 0, 0, 0, 0};
    __builtin_amdgcn_tensor_load_to_lds(g0, g1, z4, z4, z8, 0);
#else
    __builtin_amdgcn_tensor_load_to_lds(g0, g1, z4, z4, 0);
#endif
}

__device__ __forceinline__ unsigned lds_off(const void* p) {
    return (unsigned)(uintptr_t)p;   // generic shared addr: low 32 bits = LDS offset
}

// ---------------- kernel 0: bulk f32 -> f16 conversion ----------------
__global__ __launch_bounds__(256) void cvt_f16_kernel(
    const float* __restrict__ src, _Float16* __restrict__ dst, int n)
{
    const int i = (blockIdx.x * 256 + threadIdx.x) * 8;
    if (i + 8 <= n) {
        v4f x0 = *(const v4f*)(src + i);
        v4f x1 = *(const v4f*)(src + i + 4);
        v8h o;
#pragma unroll
        for (int j = 0; j < 4; ++j) { o[j] = (_Float16)x0[j]; o[j + 4] = (_Float16)x1[j]; }
        *(v8h*)(dst + i) = o;
    }
}

// ---------------- kernel 1: QKV projection GEMM, 128x128 block tile ----------------
// TDM double-buffered LDS staging; 8 waves, each wave: 2 m-tiles x 4 n-tiles.
__global__ __launch_bounds__(256) void qkv_gemm_kernel(
    const _Float16* __restrict__ Xh,   // [4096, 1024]
    const _Float16* __restrict__ Wh,   // [3, 1024, 1024]
    _Float16* __restrict__ Qh,         // [B,H,S,D] (pre-scaled by D^-0.5)
    _Float16* __restrict__ Kh,         // [B,H,S,D]
    _Float16* __restrict__ Vt)         // [B,H,D,S]
{
    // 64-half rows padded to 72 (TDM pad: 128B interval -> +16B) => conflict-free b128 reads
    __shared__ __align__(16) _Float16 XL[2][128][72];
    __shared__ __align__(16) _Float16 WL[2][128][72];

    const int tid  = threadIdx.x;
    const int lane = tid & 31, wave = tid >> 5;
    const int col  = lane & 15, half = lane >> 4;
    const int p    = blockIdx.y;                 // uniform: 0=Q,1=K,2=V
    const int mb   = (blockIdx.x >> 3) * 128;
    const int nb   = (blockIdx.x & 7) * 128;
    const int mt   = (wave & 3) * 32;            // wave's 2 m-tiles: mt, mt+16
    const int nt   = (wave >> 2) * 64;           // wave's 4 n-tiles

    const _Float16* Wp = Wh + (size_t)p * EMB * EMB;

    v8f acc[2][4] = {};

    if (tid < 32) {  // wave 0 drives the TDM pipeline
        tdm_load_2d(lds_off(&XL[0][0][0]), Xh + (size_t)mb * EMB,
                    EMB, 4096, 64, 128, EMB, 4, 3);
        tdm_load_2d(lds_off(&WL[0][0][0]), Wp + (size_t)nb * EMB,
                    EMB, EMB, 64, 128, EMB, 4, 3);
    }

    int buf = 0;
    for (int k = 0; k < EMB; k += 64, buf ^= 1) {
        if (tid < 32) __builtin_amdgcn_s_wait_tensorcnt((short)0);
        __syncthreads();                          // tiles[buf] ready; prev reads done
        if (k + 64 < EMB && tid < 32) {
            tdm_load_2d(lds_off(&XL[buf ^ 1][0][0]), Xh + (size_t)mb * EMB + k + 64,
                        EMB, 4096, 64, 128, EMB, 4, 3);
            tdm_load_2d(lds_off(&WL[buf ^ 1][0][0]), Wp + (size_t)nb * EMB + k + 64,
                        EMB, EMB, 64, 128, EMB, 4, 3);
        }
#pragma unroll
        for (int ks = 0; ks < 2; ++ks) {
            const v16h a0 = load_a16(&XL[buf][mt + col][ks * 32], half);
            const v16h a1 = load_a16(&XL[buf][mt + 16 + col][ks * 32], half);
#pragma unroll
            for (int j = 0; j < 4; ++j) {
                const v16h b = load_b16(&WL[buf][nt + j * 16 + col][ks * 32 + 16 * half]);
                acc[0][j] = WMMA_F16(a0, b, acc[0][j]);
                acc[1][j] = WMMA_F16(a1, b, acc[1][j]);
            }
        }
    }

    // epilogue: p is uniform -> scalar branches
    const float scale = (p == 0) ? 0.125f : 1.0f;
#pragma unroll
    for (int am = 0; am < 2; ++am) {
        const int gm = mb + mt + am * 16 + 8 * half;      // rows gm..gm+7
        const int bidx = gm >> 11;
        const int s0 = gm & (SQ - 1);
#pragma unroll
        for (int j = 0; j < 4; ++j) {
            const int gn = nb + nt + j * 16 + col;
            const int h = gn >> 6, d = gn & 63;
            if (p == 2) {
                v8h pk;
#pragma unroll
                for (int r = 0; r < 8; ++r) pk[r] = (_Float16)acc[am][j][r];
                *(v8h*)&Vt[(((size_t)bidx * NH + h) * HD + d) * SQ + s0] = pk;
            } else {
                _Float16* dst = (p == 0) ? Qh : Kh;
#pragma unroll
                for (int r = 0; r < 8; ++r)
                    dst[(((size_t)bidx * NH + h) * SQ + s0 + r) * HD + d] =
                        (_Float16)(acc[am][j][r] * scale);
            }
        }
    }
}

// ---------------- kernel 2: flash attention, 8 q-tiles/block sharing K/V via TDM ----------------
__global__ __launch_bounds__(256) void attn_kernel(
    const _Float16* __restrict__ Qh,   // [B,H,S,D]
    const _Float16* __restrict__ Kh,   // [B,H,S,D]
    const _Float16* __restrict__ Vt,   // [B,H,D,S]
    const float* __restrict__ mask,    // [B,S]
    float* __restrict__ Out)           // [B,S,E]
{
    __shared__ __align__(16) _Float16 ldsK[2][32][72];   // keys x d   (64->72 pad)
    __shared__ __align__(16) _Float16 ldsV[2][64][40];   // d x keys   (32->40 pad)
    __shared__ __align__(16) _Float16 plds[8][16][40];   // per-wave P transpose tile

    const int tid  = threadIdx.x;
    const int lane = tid & 31, wave = tid >> 5;
    const int col  = lane & 15, half = lane >> 4;
    const int bh   = blockIdx.x >> 4;               // 32 (b,h) pairs x 16 q-blocks
    const int qblk = blockIdx.x & 15;
    const int b    = bh >> 4, h = bh & 15;
    const int qb   = qblk * 128 + wave * 16;

    const _Float16* Qb = Qh + (size_t)bh * SQ * HD;
    const _Float16* Kb = Kh + (size_t)bh * SQ * HD;
    const _Float16* Vb = Vt + (size_t)bh * HD * SQ;
    const float* mp    = mask + (size_t)b * SQ;

    const _Float16* qrow = Qb + (size_t)(qb + col) * HD;
    const v16h aQ0 = load_a16(qrow, half);
    const v16h aQ1 = load_a16(qrow + 32, half);

    float mrow[8], lrow[8];
    v8f acc[4] = {};
#pragma unroll
    for (int r = 0; r < 8; ++r) { mrow[r] = -INFINITY; lrow[r] = 0.0f; }

    if (tid < 32) {  // prologue: stage first K/V tile pair
        tdm_load_2d(lds_off(&ldsK[0][0][0]), Kb, HD, SQ, HD, 32, HD, 4, 3);
        tdm_load_2d(lds_off(&ldsV[0][0][0]), Vb, SQ, HD, 32, HD, SQ, 3, 3);
    }

    int buf = 0;
    for (int kb = 0; kb < SQ; kb += 32, buf ^= 1) {
        if (tid < 32) __builtin_amdgcn_s_wait_tensorcnt((short)0);
        __syncthreads();                            // tiles[buf] ready; prev iter fully done
        if (kb + 32 < SQ && tid < 32) {
            tdm_load_2d(lds_off(&ldsK[buf ^ 1][0][0]), Kb + (size_t)(kb + 32) * HD,
                        HD, SQ, HD, 32, HD, 4, 3);
            tdm_load_2d(lds_off(&ldsV[buf ^ 1][0][0]), Vb + (kb + 32),
                        SQ, HD, 32, HD, SQ, 3, 3);
        }

        // ---- scores from LDS K tile ----
        const _Float16* kr0 = &ldsK[buf][col][0];
        const _Float16* kr1 = &ldsK[buf][col + 16][0];
        v8f s0 = {}, s1 = {};
        s0 = WMMA_F16(aQ0, load_b16(kr0 + 16 * half), s0);
        s0 = WMMA_F16(aQ1, load_b16(kr0 + 32 + 16 * half), s0);
        s1 = WMMA_F16(aQ0, load_b16(kr1 + 16 * half), s1);
        s1 = WMMA_F16(aQ1, load_b16(kr1 + 32 + 16 * half), s1);

        const float pen0 = 1e30f * (1.0f - mp[kb + col]);
        const float pen1 = 1e30f * (1.0f - mp[kb + 16 + col]);

        // ---- online softmax (rows live in 16-lane halves) ----
#pragma unroll
        for (int r = 0; r < 8; ++r) {
            float v0 = s0[r] - pen0;
            float v1 = s1[r] - pen1;
            float mx = fmaxf(v0, v1);
#pragma unroll
            for (int off = 8; off; off >>= 1) mx = fmaxf(mx, __shfl_xor(mx, off, 16));
            const float mnew = fmaxf(mrow[r], mx);
            const float p0 = __expf(v0 - mnew);
            const float p1 = __expf(v1 - mnew);
            float rs = p0 + p1;
#pragma unroll
            for (int off = 8; off; off >>= 1) rs += __shfl_xor(rs, off, 16);
            const float corr = __expf(mrow[r] - mnew);
            mrow[r] = mnew;
            lrow[r] = lrow[r] * corr + rs;
            acc[0][r] *= corr; acc[1][r] *= corr;
            acc[2][r] *= corr; acc[3][r] *= corr;
            plds[wave][r + 8 * half][col]      = (_Float16)p0;
            plds[wave][r + 8 * half][16 + col] = (_Float16)p1;
        }
        __syncthreads();                            // P tile written (also fences LDS)

        const v16h aP = load_a16(&plds[wave][col][0], half);

        // ---- P @ V from LDS V tile ----
#pragma unroll
        for (int ch = 0; ch < 4; ++ch) {
            const _Float16* vptr = &ldsV[buf][ch * 16 + col][16 * half];
            acc[ch] = WMMA_F16(aP, load_b16(vptr), acc[ch]);
        }
    }

    // ---- epilogue: normalize, scatter to [B,S,E] ----
#pragma unroll
    for (int r = 0; r < 8; ++r) {
        const float inv = 1.0f / lrow[r];
        const int q = qb + r + 8 * half;
        const size_t base = ((size_t)b * SQ + q) * EMB + (size_t)h * HD;
#pragma unroll
        for (int ch = 0; ch < 4; ++ch)
            Out[base + ch * 16 + col] = acc[ch][r] * inv;
    }
}

extern "C" void kernel_launch(void* const* d_in, const int* in_sizes, int n_in,
                              void* d_out, int out_size, void* d_ws, size_t ws_size,
                              hipStream_t stream) {
    const float* X    = (const float*)d_in[0];   // [B,S,E]
    const float* mask = (const float*)d_in[1];   // [B,S]
    const float* Wq   = (const float*)d_in[2];   // [E,E]
    const float* Wk   = (const float*)d_in[3];
    const float* Wv   = (const float*)d_in[4];
    float* out = (float*)d_out;

    const size_t nX = (size_t)BB * SQ * EMB;         // 4 Mi
    const size_t nW = (size_t)EMB * EMB;             // 1 Mi
    const size_t nQKV = (size_t)BB * NH * SQ * HD;   // 4 Mi

    _Float16* Xh = (_Float16*)d_ws;
    _Float16* Wh = Xh + nX;
    _Float16* Qh = Wh + 3 * nW;
    _Float16* Kh = Qh + nQKV;
    _Float16* Vt = Kh + nQKV;

    cvt_f16_kernel<<<(int)(nX / 8 / 256), 256, 0, stream>>>(X, Xh, (int)nX);
    cvt_f16_kernel<<<(int)(nW / 8 / 256), 256, 0, stream>>>(Wq, Wh + 0 * nW, (int)nW);
    cvt_f16_kernel<<<(int)(nW / 8 / 256), 256, 0, stream>>>(Wk, Wh + 1 * nW, (int)nW);
    cvt_f16_kernel<<<(int)(nW / 8 / 256), 256, 0, stream>>>(Wv, Wh + 2 * nW, (int)nW);

    dim3 ggrid(256, 3);                              // 32 m-blocks x 8 n-blocks, 3 projections
    qkv_gemm_kernel<<<ggrid, 256, 0, stream>>>(Xh, Wh, Qh, Kh, Vt);

    attn_kernel<<<BB * NH * (SQ / 128), 256, 0, stream>>>(Qh, Kh, Vt, mask, out);
}
